// ImprovedGRUCell_42322607735488
// MI455X (gfx1250) — matrix operand order repaired and checked by reading, not compiled, not run on hardware
//
#include <hip/hip_runtime.h>
#include <hip/hip_bf16.h>

// ---------------------------------------------------------------------------
// Fused "improved GRU cell" for gfx1250 (MI455X).
// bf16 WMMA (v_wmma_f32_16x16x32_bf16) for all six GEMMs, f32 accumulate.
// One workgroup = 32 batch rows x full 256 hidden columns (softmax needs the
// whole row), 8 wave32s, each wave owns a 32-column slab.
// ---------------------------------------------------------------------------

typedef __attribute__((ext_vector_type(16))) __bf16 v16bf;
typedef __attribute__((ext_vector_type(8)))  __bf16 v8bf;
typedef __attribute__((ext_vector_type(8)))  float  v8f;

#define HID 256
#define TILE_M 32
#define XH_STRIDE 520   // 512 + 8 halfs: rows 16B-aligned, 1040B = 260 dwords -> bank-conflict-free TR reads
#define ATT_STRIDE 264  // 256 + 8 halfs
#define S_STRIDE  264
#define HS_STRIDE 264

__device__ __forceinline__ float fast_sigmoid(float x) { return 1.0f / (1.0f + __expf(-x)); }
__device__ __forceinline__ float fast_tanh(float x)    { return 2.0f / (1.0f + __expf(-2.0f * x)) - 1.0f; }

__device__ __forceinline__ v8f wmma_bf16(v16bf a, v16bf b, v8f c) {
    // (neg_a, A, neg_b, B, c_mod, C, reuse_a, reuse_b)
    return __builtin_amdgcn_wmma_f32_16x16x32_bf16(false, a, false, b, (short)0, c, false, false);
}

// A-matrix fragment (16x32 bf16, M x K) from a row-major bf16 LDS tile.
// Layout: lane L<16 holds row L, K = k0..k0+15;
//         lane L+16 holds row L, K = k0+16..k0+31.
__device__ __forceinline__ v16bf lds_a_frag(const __bf16* base, int stride, int k0, int lane) {
    const int r   = lane & 15;
    const int sel = (lane >> 4) & 1;
    const __bf16* p = base + r * stride + k0 + sel * 16;
    v8bf lo = *(const v8bf*)(p);        // 16B -> ds_load_b128
    v8bf hi = *(const v8bf*)(p + 8);    // 16B -> ds_load_b128
    v16bf a;
#pragma unroll
    for (int i = 0; i < 8; ++i) { a[i] = lo[i]; a[i + 8] = hi[i]; }
    return a;
}

// B-matrix fragment (32x16 bf16, K x N) for out = act @ W^T:
// B column n == row n of W.  lane<16: col n0+lane, K=k0..k0+15;
// lane>=16: col n0+(lane-16), K=k0+16..k0+31. One contiguous 32B load.
__device__ __forceinline__ v16bf glb_b_frag(const __bf16* __restrict__ W, int n0, int k0, int lane) {
    const int n = n0 + (lane & 15);
    const int k = k0 + ((lane >> 4) & 1) * 16;
    return *(const v16bf*)(W + n * HID + k);
}

__global__ void __launch_bounds__(256)
gru_fused_kernel(const float* __restrict__ x, const float* __restrict__ h,
                 const float* __restrict__ b_z, const float* __restrict__ v_a,
                 const float* __restrict__ b_h,
                 const __bf16* __restrict__ Wz, const __bf16* __restrict__ Uz,
                 const __bf16* __restrict__ Wa, const __bf16* __restrict__ Ua,
                 const __bf16* __restrict__ Wh, const __bf16* __restrict__ Uh,
                 float* __restrict__ out)
{
    __shared__ __align__(32) __bf16 sXH[TILE_M * XH_STRIDE];   // [32][512] x|h bf16 (A source)
    __shared__ __align__(32) __bf16 sATT[TILE_M * ATT_STRIDE]; // [32][256] attended_h bf16 (A source)
    __shared__ __align__(16) float  sS[TILE_M * S_STRIDE];     // scores -> exp values
    __shared__ __align__(16) float  sH[TILE_M * HS_STRIDE];    // f32 copy of h_prev tile
    __shared__ float sRedA[TILE_M * 8];
    __shared__ float sRedB[TILE_M * 8];

    const int tid     = threadIdx.x;
    const int lane    = tid & 31;
    const int wv      = tid >> 5;
    const int rowBase = blockIdx.x * TILE_M;
    const int colBase = wv * 32;
    const int hl      = (lane >> 4) & 1;

    // ---------------- stage x,h tile: global f32 -> LDS bf16 (+ f32 h copy) ----
    {
        const float4* xg = (const float4*)(x) + rowBase * (HID / 4);
        const float4* hg = (const float4*)(h) + rowBase * (HID / 4);
#pragma unroll
        for (int i = 0; i < 2; ++i) {
            const int idx = i * 256 + tid;      // 512 16B chunks, fully coalesced
            const int rr  = idx >> 4;
            const int c4  = idx & 15;
            const float4 vx = xg[rr * (HID / 4) + c4];
            const float4 vh = hg[rr * (HID / 4) + c4];
            const int cc = c4 * 4;
            __bf16* px = &sXH[rr * XH_STRIDE + cc];
            px[0] = (__bf16)vx.x; px[1] = (__bf16)vx.y; px[2] = (__bf16)vx.z; px[3] = (__bf16)vx.w;
            __bf16* ph = &sXH[rr * XH_STRIDE + HID + cc];
            ph[0] = (__bf16)vh.x; ph[1] = (__bf16)vh.y; ph[2] = (__bf16)vh.z; ph[3] = (__bf16)vh.w;
            float* pf = &sH[rr * HS_STRIDE + cc];
            pf[0] = vh.x; pf[1] = vh.y; pf[2] = vh.z; pf[3] = vh.w;
        }
    }
    __syncthreads();

    const v8f vzero = {0.f, 0.f, 0.f, 0.f, 0.f, 0.f, 0.f, 0.f};
    v8f accZ[2][2], accA[2][2];
#pragma unroll
    for (int mi = 0; mi < 2; ++mi)
#pragma unroll
        for (int ni = 0; ni < 2; ++ni) { accZ[mi][ni] = vzero; accA[mi][ni] = vzero; }

    // ---------------- phase 1: zpre = x@Wz^T + h@Uz^T ; apre likewise --------
#pragma unroll
    for (int it = 0; it < 8; ++it) {           // x part (K = 0..255)
        const int kk = it * 32;
        v16bf a0  = lds_a_frag(sXH,                  XH_STRIDE, kk, lane);
        v16bf a1  = lds_a_frag(sXH + 16 * XH_STRIDE, XH_STRIDE, kk, lane);
        v16bf bz0 = glb_b_frag(Wz, colBase,      kk, lane);
        v16bf bz1 = glb_b_frag(Wz, colBase + 16, kk, lane);
        v16bf ba0 = glb_b_frag(Wa, colBase,      kk, lane);
        v16bf ba1 = glb_b_frag(Wa, colBase + 16, kk, lane);
        accZ[0][0] = wmma_bf16(a0, bz0, accZ[0][0]);
        accZ[0][1] = wmma_bf16(a0, bz1, accZ[0][1]);
        accZ[1][0] = wmma_bf16(a1, bz0, accZ[1][0]);
        accZ[1][1] = wmma_bf16(a1, bz1, accZ[1][1]);
        accA[0][0] = wmma_bf16(a0, ba0, accA[0][0]);
        accA[0][1] = wmma_bf16(a0, ba1, accA[0][1]);
        accA[1][0] = wmma_bf16(a1, ba0, accA[1][0]);
        accA[1][1] = wmma_bf16(a1, ba1, accA[1][1]);
    }
#pragma unroll
    for (int it = 0; it < 8; ++it) {           // h part (K = 256..511 of xh tile)
        const int kk = it * 32;
        v16bf a0  = lds_a_frag(sXH,                  XH_STRIDE, HID + kk, lane);
        v16bf a1  = lds_a_frag(sXH + 16 * XH_STRIDE, XH_STRIDE, HID + kk, lane);
        v16bf bz0 = glb_b_frag(Uz, colBase,      kk, lane);
        v16bf bz1 = glb_b_frag(Uz, colBase + 16, kk, lane);
        v16bf ba0 = glb_b_frag(Ua, colBase,      kk, lane);
        v16bf ba1 = glb_b_frag(Ua, colBase + 16, kk, lane);
        accZ[0][0] = wmma_bf16(a0, bz0, accZ[0][0]);
        accZ[0][1] = wmma_bf16(a0, bz1, accZ[0][1]);
        accZ[1][0] = wmma_bf16(a1, bz0, accZ[1][0]);
        accZ[1][1] = wmma_bf16(a1, bz1, accZ[1][1]);
        accA[0][0] = wmma_bf16(a0, ba0, accA[0][0]);
        accA[0][1] = wmma_bf16(a0, ba1, accA[0][1]);
        accA[1][0] = wmma_bf16(a1, ba0, accA[1][0]);
        accA[1][1] = wmma_bf16(a1, ba1, accA[1][1]);
    }

    // ---------------- phase 2: z = sigmoid(.), scores = tanh(.)*v_a ----------
    // C/D layout: vgpr i, lane<16 -> (M=i, N=lane); lane>=16 -> (M=8+i, N=lane-16)
#pragma unroll
    for (int mi = 0; mi < 2; ++mi) {
#pragma unroll
        for (int ni = 0; ni < 2; ++ni) {
            const int col = colBase + ni * 16 + (lane & 15);
            const float bz = b_z[col];
            const float va = v_a[col];
            const int rb = mi * 16 + hl * 8;
#pragma unroll
            for (int i = 0; i < 8; ++i) {
                accZ[mi][ni][i] = fast_sigmoid(accZ[mi][ni][i] + bz);    // keep z in regs
                sS[(rb + i) * S_STRIDE + col] = fast_tanh(accA[mi][ni][i]) * va;
            }
        }
    }
    __syncthreads();

    // ---------------- softmax over hidden dim + attended_h = w * h_prev ------
    {
        const int r = tid >> 3;        // row 0..31
        const int g = tid & 7;         // 8 threads per row, 32 cols each
        float* srow = sS + r * S_STRIDE + g * 32;
        float m = -3.4e38f;
#pragma unroll
        for (int j = 0; j < 32; ++j) m = fmaxf(m, srow[j]);
        sRedA[r * 8 + g] = m;
        __syncthreads();
        float rowmax = sRedA[r * 8 + 0];
#pragma unroll
        for (int j = 1; j < 8; ++j) rowmax = fmaxf(rowmax, sRedA[r * 8 + j]);
        float psum = 0.f;
#pragma unroll
        for (int j = 0; j < 32; ++j) {
            const float e = __expf(srow[j] - rowmax);
            srow[j] = e;
            psum += e;
        }
        sRedB[r * 8 + g] = psum;
        __syncthreads();
        float rowsum = 0.f;
#pragma unroll
        for (int j = 0; j < 8; ++j) rowsum += sRedB[r * 8 + j];
        const float inv = 1.0f / rowsum;
        const float* hrow = sH + r * HS_STRIDE + g * 32;
        __bf16* arow = sATT + r * ATT_STRIDE + g * 32;
#pragma unroll
        for (int j = 0; j < 32; ++j) arow[j] = (__bf16)(srow[j] * inv * hrow[j]);
    }
    __syncthreads();

    // ---------------- phase 3: hpre = x@Wh^T + attended@Uh^T -----------------
    v8f accH[2][2];
#pragma unroll
    for (int mi = 0; mi < 2; ++mi)
#pragma unroll
        for (int ni = 0; ni < 2; ++ni) accH[mi][ni] = vzero;

#pragma unroll
    for (int it = 0; it < 8; ++it) {           // x part
        const int kk = it * 32;
        v16bf a0 = lds_a_frag(sXH,                  XH_STRIDE, kk, lane);
        v16bf a1 = lds_a_frag(sXH + 16 * XH_STRIDE, XH_STRIDE, kk, lane);
        v16bf b0 = glb_b_frag(Wh, colBase,      kk, lane);
        v16bf b1 = glb_b_frag(Wh, colBase + 16, kk, lane);
        accH[0][0] = wmma_bf16(a0, b0, accH[0][0]);
        accH[0][1] = wmma_bf16(a0, b1, accH[0][1]);
        accH[1][0] = wmma_bf16(a1, b0, accH[1][0]);
        accH[1][1] = wmma_bf16(a1, b1, accH[1][1]);
    }
#pragma unroll
    for (int it = 0; it < 8; ++it) {           // attended part
        const int kk = it * 32;
        v16bf a0 = lds_a_frag(sATT,                   ATT_STRIDE, kk, lane);
        v16bf a1 = lds_a_frag(sATT + 16 * ATT_STRIDE, ATT_STRIDE, kk, lane);
        v16bf b0 = glb_b_frag(Uh, colBase,      kk, lane);
        v16bf b1 = glb_b_frag(Uh, colBase + 16, kk, lane);
        accH[0][0] = wmma_bf16(a0, b0, accH[0][0]);
        accH[0][1] = wmma_bf16(a0, b1, accH[0][1]);
        accH[1][0] = wmma_bf16(a1, b0, accH[1][0]);
        accH[1][1] = wmma_bf16(a1, b1, accH[1][1]);
    }

    // ---------------- phase 4: h_t = (1-z)*h_prev + z*tanh(hpre + b_h) -------
#pragma unroll
    for (int mi = 0; mi < 2; ++mi) {
#pragma unroll
        for (int ni = 0; ni < 2; ++ni) {
            const int col = colBase + ni * 16 + (lane & 15);
            const float bh = b_h[col];
            const int rb = mi * 16 + hl * 8;
#pragma unroll
            for (int i = 0; i < 8; ++i) {
                const float ht = fast_tanh(accH[mi][ni][i] + bh);
                const float hp = sH[(rb + i) * HS_STRIDE + col];
                const float z  = accZ[mi][ni][i];
                out[(size_t)(rowBase + rb + i) * HID + col] = hp + z * (ht - hp);
            }
        }
    }
}

// One-shot weight conversion f32 -> bf16 into workspace (L2-resident afterwards).
__global__ void cvt_weights_kernel(const float* __restrict__ s0, const float* __restrict__ s1,
                                   const float* __restrict__ s2, const float* __restrict__ s3,
                                   const float* __restrict__ s4, const float* __restrict__ s5,
                                   __bf16* __restrict__ dst)
{
    const int i = blockIdx.x * 256 + threadIdx.x;          // 0..65535
    const int m = blockIdx.y;                              // which matrix
    const float* s = (m == 0) ? s0 : (m == 1) ? s1 : (m == 2) ? s2
                   : (m == 3) ? s3 : (m == 4) ? s4 : s5;
    dst[(size_t)m * (HID * HID) + i] = (__bf16)s[i];
}

extern "C" void kernel_launch(void* const* d_in, const int* in_sizes, int n_in,
                              void* d_out, int out_size, void* d_ws, size_t ws_size,
                              hipStream_t stream) {
    const float* x   = (const float*)d_in[0];
    const float* h   = (const float*)d_in[1];
    const float* W_z = (const float*)d_in[2];
    const float* U_z = (const float*)d_in[3];
    const float* b_z = (const float*)d_in[4];
    const float* W_a = (const float*)d_in[5];
    const float* U_a = (const float*)d_in[6];
    const float* v_a = (const float*)d_in[7];
    const float* W_h = (const float*)d_in[8];
    const float* U_h = (const float*)d_in[9];
    const float* b_h = (const float*)d_in[10];

    __bf16* wbf = (__bf16*)d_ws;   // 6 * 256*256 bf16 = 768 KB
    cvt_weights_kernel<<<dim3(256, 6), 256, 0, stream>>>(W_z, U_z, W_a, U_a, W_h, U_h, wbf);

    const int B = in_sizes[0] / HID;          // 65536
    const int nBlocks = B / TILE_M;           // 2048
    gru_fused_kernel<<<nBlocks, 256, 0, stream>>>(
        x, h, b_z, v_a, b_h,
        wbf + 0 * (HID * HID),   // Wz
        wbf + 1 * (HID * HID),   // Uz
        wbf + 2 * (HID * HID),   // Wa
        wbf + 3 * (HID * HID),   // Ua
        wbf + 4 * (HID * HID),   // Wh
        wbf + 5 * (HID * HID),   // Uh
        (float*)d_out);
}
